// Attention_32916629357206
// MI455X (gfx1250) — matrix operand order repaired and checked by reading, not compiled, not run on hardware
//
#include <hip/hip_runtime.h>
#include <hip/hip_bf16.h>
#include <cstdint>

#define B_   2
#define S_   2048
#define H_   2048
#define NH_  32
#define NKV_ 8
#define HD_  64
#define NEGV (-1000000000.0f)

typedef __bf16 bf16_t;
typedef __attribute__((ext_vector_type(8)))  bf16_t v8bf;
typedef __attribute__((ext_vector_type(16))) bf16_t v16bf;
typedef __attribute__((ext_vector_type(8)))  float  v8f;

union Frag {
  v16bf v;
  v8bf  h[2];
  bf16_t e[16];
};

__device__ __forceinline__ v8bf load8(const bf16_t* p) {
  return *reinterpret_cast<const v8bf*>(p);
}

__device__ __forceinline__ v8f wmma_bf16(v16bf a, v16bf b, v8f c) {
  // D = A(16x32 bf16) * B(32x16 bf16) + C(16x16 f32)
  return __builtin_amdgcn_wmma_f32_16x16x32_bf16(false, a, false, b, (short)0, c,
                                                 false, false);
}

__device__ __forceinline__ void wait_ds0() {
#if __has_builtin(__builtin_amdgcn_s_wait_dscnt)
  __builtin_amdgcn_s_wait_dscnt(0);
#else
  asm volatile("s_wait_dscnt 0" ::: "memory");
#endif
}

// ---------------------------------------------------------------- convert
__global__ void cvt_f32_bf16(const float* __restrict__ in, bf16_t* __restrict__ out,
                             int n) {
  int i = blockIdx.x * blockDim.x + threadIdx.x;
  if (i < n) out[i] = (bf16_t)in[i];
}

// ---------------------------------------------------------------- GEMM
// out[m,n] = sum_k A[m,k] * W[n,k] + bias[n]
// A: MxK bf16 row-major, W: NxK bf16 row-major.
// Block = 128 threads = 4 waves. Wave computes a 64(M) x 64(N) tile
// (4 A-frags x 4 B-frags -> 16 WMMAs per K-step of 32).
// grid = (M/64, N/256)
// TRANS: store transposed as out[(b, n, s)] with s contiguous (for V^T),
//        vectorized 16B stores since a lane's 8 acc elements are 8 consecutive rows.
template <typename OutT, bool TRANS = false>
__global__ void __launch_bounds__(128)
gemm_bias_kernel(const bf16_t* __restrict__ A, const bf16_t* __restrict__ W,
                 const float* __restrict__ bias, OutT* __restrict__ out,
                 int M, int N, int K) {
  const int lane = threadIdx.x & 31;
  const int wave = threadIdx.x >> 5;
  const int l16  = lane & 15;
  const int half = lane >> 4;
  const int m0 = blockIdx.x * 64;
  const int n0 = blockIdx.y * 256 + wave * 64;
  (void)M;

  v8f acc[4][4] = {};

  const bf16_t* arow[4];
#pragma unroll
  for (int i = 0; i < 4; ++i) arow[i] = A + (size_t)(m0 + i * 16 + l16) * K;

  for (int kk = 0; kk < K; kk += 32) {
    // A fragments (16x32): lane = M row; 16 contiguous K {half*16 .. +15}
    Frag a[4];
    const int ka = kk + half * 16;
#pragma unroll
    for (int i = 0; i < 4; ++i) {
      a[i].h[0] = load8(arow[i] + ka);
      a[i].h[1] = load8(arow[i] + ka + 8);
    }
    const int kb = kk + half * 16;  // B frag: lane = N col, 16 contiguous K
#pragma unroll
    for (int j = 0; j < 4; ++j) {
      Frag bfr;
      const bf16_t* wrow = W + (size_t)(n0 + j * 16 + l16) * K + kb;
      bfr.h[0] = load8(wrow);
      bfr.h[1] = load8(wrow + 8);
#pragma unroll
      for (int i = 0; i < 4; ++i) acc[i][j] = wmma_bf16(a[i].v, bfr.v, acc[i][j]);
    }
  }

#pragma unroll
  for (int j = 0; j < 4; ++j) {
    const int ncol = n0 + j * 16 + l16;
    const float bv = bias[ncol];
#pragma unroll
    for (int i = 0; i < 4; ++i) {
      const int row0 = m0 + i * 16 + half * 8;  // lane's 8 rows: row0 .. row0+7
      if constexpr (TRANS) {
        // out layout: [b][n][s], s contiguous; rows row0..row0+7 share b.
        const int bb = row0 >> 11;        // row / S_
        const int ss = row0 & (S_ - 1);
        v8bf pack;
#pragma unroll
        for (int r = 0; r < 8; ++r) pack[r] = (bf16_t)(acc[i][j][r] + bv);
        *reinterpret_cast<v8bf*>(out + ((size_t)bb * N + ncol) * S_ + ss) = pack;
      } else {
#pragma unroll
        for (int r = 0; r < 8; ++r)
          out[(size_t)(row0 + r) * N + ncol] = (OutT)(acc[i][j][r] + bv);
      }
    }
  }
}

// ---------------------------------------------------------------- RoPE (in place)
__global__ void rope_kernel(bf16_t* __restrict__ buf, int n_heads, int total) {
  int idx = blockIdx.x * blockDim.x + threadIdx.x;
  if (idx >= total) return;
  int i    = idx & 31;        // rotation pair index 0..31
  int rest = idx >> 5;
  int head = rest % n_heads;
  int tok  = rest / n_heads;  // b*S + s
  int pos  = tok & (S_ - 1);
  float ang = (float)pos * __powf(10000.0f, -(float)i * (1.0f / 32.0f));
  float sn, cs;
  __sincosf(ang, &sn, &cs);
  bf16_t* p = buf + ((size_t)tok * n_heads + head) * HD_;
  float x1 = (float)p[i];
  float x2 = (float)p[i + 32];
  p[i]      = (bf16_t)(x1 * cs - x2 * sn);
  p[i + 32] = (bf16_t)(x2 * cs + x1 * sn);
}

// ---------------------------------------------------------------- attention
// Q: (B*S, NH, 64) bf16, K: (B*S, NKV, 64) bf16, VT: (B, NKV, 64, S) bf16,
// O: (B*S, NH, 64) bf16
// grid = (B*NH, S/64); block = 128 (4 waves, each owning a 16-query tile)
__global__ void __launch_bounds__(128)
attn_kernel(const bf16_t* __restrict__ Q, const bf16_t* __restrict__ Kc,
            const bf16_t* __restrict__ VT, bf16_t* __restrict__ O) {
  __shared__ bf16_t pbuf[4][16][40];  // padded: 40*2B = 80B row stride
  const int lane = threadIdx.x & 31;
  const int wave = threadIdx.x >> 5;
  const int l16  = lane & 15;
  const int half = lane >> 4;
  const int bh = blockIdx.x;
  const int b  = bh >> 5;
  const int h  = bh & 31;
  const int kvh = h >> 2;  // GQA: 4 query heads per KV head
  const int q0 = blockIdx.y * 64 + wave * 16;
  const float scale = 0.125f;  // 1/sqrt(64)

  // Q tile as two 16x32 A-fragments (d 0..31 and 32..63)
  Frag qa[2];
  {
    const bf16_t* qrow = Q + ((size_t)(b * S_ + q0 + l16) * NH_ + h) * HD_;
#pragma unroll
    for (int d = 0; d < 2; ++d) {
      const int ka = d * 32 + half * 16;
      qa[d].h[0] = load8(qrow + ka);
      qa[d].h[1] = load8(qrow + ka + 8);
    }
  }

  v8f o[4] = {};
  float mrow[8], lrow[8];
#pragma unroll
  for (int r = 0; r < 8; ++r) { mrow[r] = -3.0e38f; lrow[r] = 0.0f; }

  const bf16_t* vthead = VT + (size_t)(b * NKV_ + kvh) * HD_ * S_;

  const int nblk = (q0 + 16 + 31) >> 5;  // causal: keys 0 .. q0+15
  for (int kb = 0; kb < nblk; ++kb) {
    const int k0 = kb * 32;

    // scores: two 16x16 tiles (keys lo/hi), each 2 WMMAs over d
    v8f sc[2] = {};
#pragma unroll
    for (int t = 0; t < 2; ++t) {
      const bf16_t* krow =
          Kc + ((size_t)(b * S_ + k0 + t * 16 + l16) * NKV_ + kvh) * HD_;
#pragma unroll
      for (int d = 0; d < 2; ++d) {
        Frag kf;  // B frag: lane = key col, 16 contiguous d values
        kf.h[0] = load8(krow + d * 32 + half * 16);
        kf.h[1] = load8(krow + d * 32 + half * 16 + 8);
        sc[t] = wmma_bf16(qa[d].v, kf.v, sc[t]);
      }
    }

    // scale + causal mask (matches reference: s*scale + NEG)
    const int qr0 = q0 + half * 8;
#pragma unroll
    for (int t = 0; t < 2; ++t) {
      const int key = k0 + t * 16 + l16;
#pragma unroll
      for (int r = 0; r < 8; ++r) {
        float vv = sc[t][r] * scale;
        if (key > qr0 + r) vv += NEGV;
        sc[t][r] = vv;
      }
    }

    // online softmax; row = 16 lanes of one half -> shfl_xor 1,2,4,8
#pragma unroll
    for (int r = 0; r < 8; ++r) {
      float mx = fmaxf(sc[0][r], sc[1][r]);
      mx = fmaxf(mx, __shfl_xor(mx, 1, 32));
      mx = fmaxf(mx, __shfl_xor(mx, 2, 32));
      mx = fmaxf(mx, __shfl_xor(mx, 4, 32));
      mx = fmaxf(mx, __shfl_xor(mx, 8, 32));
      const float newm  = fmaxf(mrow[r], mx);
      const float alpha = __expf(mrow[r] - newm);
      float p0 = __expf(sc[0][r] - newm);
      float p1 = __expf(sc[1][r] - newm);
      sc[0][r] = p0;
      sc[1][r] = p1;
      float rs = p0 + p1;
      rs += __shfl_xor(rs, 1, 32);
      rs += __shfl_xor(rs, 2, 32);
      rs += __shfl_xor(rs, 4, 32);
      rs += __shfl_xor(rs, 8, 32);
      lrow[r] = lrow[r] * alpha + rs;
      mrow[r] = newm;
#pragma unroll
      for (int dd = 0; dd < 4; ++dd) o[dd][r] *= alpha;
    }

    // P: C-layout -> LDS -> reread as 16x32 A-fragment (wave-private)
#pragma unroll
    for (int t = 0; t < 2; ++t)
#pragma unroll
      for (int r = 0; r < 8; ++r)
        pbuf[wave][r + half * 8][t * 16 + l16] = (bf16_t)sc[t][r];
    wait_ds0();
    Frag pf;
    pf.h[0] = load8(&pbuf[wave][l16][half * 16]);
    pf.h[1] = load8(&pbuf[wave][l16][half * 16 + 8]);

    // accumulate P(16x32) @ V(32x64) as 4 WMMAs; V^T rows are contiguous in s
#pragma unroll
    for (int dd = 0; dd < 4; ++dd) {
      Frag vf;  // B frag: lane = d col, elements = 16 contiguous keys
      const bf16_t* vtrow = vthead + (size_t)(dd * 16 + l16) * S_ + k0 + half * 16;
      vf.h[0] = load8(vtrow);
      vf.h[1] = load8(vtrow + 8);
      o[dd] = wmma_bf16(pf.v, vf.v, o[dd]);
    }
  }

  // normalize and store (b, s, head, d) row-major -> feeds O-proj GEMM
#pragma unroll
  for (int dd = 0; dd < 4; ++dd) {
#pragma unroll
    for (int r = 0; r < 8; ++r) {
      const float val = o[dd][r] / lrow[r];
      O[((size_t)(b * S_ + q0 + r + half * 8) * NH_ + h) * HD_ + dd * 16 + l16] =
          (bf16_t)val;
    }
  }
}

// ---------------------------------------------------------------- launch
extern "C" void kernel_launch(void* const* d_in, const int* in_sizes, int n_in,
                              void* d_out, int out_size, void* d_ws, size_t ws_size,
                              hipStream_t stream) {
  const float* hs = (const float*)d_in[0];
  // d_in[1] = attention_mask (causal; computed analytically), d_in[2] = position_ids (arange)
  const float* Wq = (const float*)d_in[3];
  const float* bq = (const float*)d_in[4];
  const float* Wk = (const float*)d_in[5];
  const float* bk = (const float*)d_in[6];
  const float* Wv = (const float*)d_in[7];
  const float* bv = (const float*)d_in[8];
  const float* Wo = (const float*)d_in[9];
  const float* bo = (const float*)d_in[10];
  float* out = (float*)d_out;
  (void)in_sizes; (void)n_in; (void)out_size; (void)ws_size;

  const int M   = B_ * S_;     // 4096
  const int K   = H_;          // 2048
  const int Nq  = NH_ * HD_;   // 2048
  const int Nkv = NKV_ * HD_;  // 512

  size_t off = 0;
  auto carve = [&](size_t bytes) -> void* {
    void* p = (char*)d_ws + off;
    off += (bytes + 255) & ~(size_t)255;
    return p;
  };
  bf16_t* hsb = (bf16_t*)carve((size_t)M * K * 2);
  bf16_t* wqb = (bf16_t*)carve((size_t)Nq * K * 2);
  bf16_t* wkb = (bf16_t*)carve((size_t)Nkv * K * 2);
  bf16_t* wvb = (bf16_t*)carve((size_t)Nkv * K * 2);
  bf16_t* wob = (bf16_t*)carve((size_t)H_ * K * 2);
  bf16_t* qb  = (bf16_t*)carve((size_t)M * Nq * 2);
  bf16_t* kbf = (bf16_t*)carve((size_t)M * Nkv * 2);
  bf16_t* vtb = (bf16_t*)carve((size_t)M * Nkv * 2);  // V^T: (B, NKV, 64, S)
  bf16_t* ob  = (bf16_t*)carve((size_t)M * Nq * 2);

  auto cvt = [&](const float* src, bf16_t* dst, size_t n) {
    cvt_f32_bf16<<<dim3((unsigned)((n + 255) / 256)), 256, 0, stream>>>(src, dst,
                                                                        (int)n);
  };
  cvt(hs, hsb, (size_t)M * K);
  cvt(Wq, wqb, (size_t)Nq * K);
  cvt(Wk, wkb, (size_t)Nkv * K);
  cvt(Wv, wvb, (size_t)Nkv * K);
  cvt(Wo, wob, (size_t)H_ * K);

  gemm_bias_kernel<bf16_t>
      <<<dim3(M / 64, Nq / 256), 128, 0, stream>>>(hsb, wqb, bq, qb, M, Nq, K);
  gemm_bias_kernel<bf16_t>
      <<<dim3(M / 64, Nkv / 256), 128, 0, stream>>>(hsb, wkb, bk, kbf, M, Nkv, K);
  gemm_bias_kernel<bf16_t, true>
      <<<dim3(M / 64, Nkv / 256), 128, 0, stream>>>(hsb, wvb, bv, vtb, M, Nkv, K);

  const int qpairs = M * NH_ * 32;
  const int kpairs = M * NKV_ * 32;
  rope_kernel<<<dim3((qpairs + 255) / 256), 256, 0, stream>>>(qb, NH_, qpairs);
  rope_kernel<<<dim3((kpairs + 255) / 256), 256, 0, stream>>>(kbf, NKV_, kpairs);

  attn_kernel<<<dim3(B_ * NH_, S_ / 64), 128, 0, stream>>>(qb, kbf, vtb, ob);

  gemm_bias_kernel<float>
      <<<dim3(M / 64, Nq / 256), 128, 0, stream>>>(ob, wob, bo, out, M, H_, K);
}
